// FMActionDistHead_54262616817936
// MI455X (gfx1250) — matrix operand order repaired and checked by reading, not compiled, not run on hardware
//
#include <hip/hip_runtime.h>

typedef __attribute__((ext_vector_type(16))) _Float16 v16h;
typedef __attribute__((ext_vector_type(8)))  _Float16 v8h;
typedef __attribute__((ext_vector_type(8)))  float    v8f;

#define B_TOT   32768
#define A_DIM   8
#define F_DIM   256
#define H_DIM   256
#define NSTEPS  50
#define IN_DIM  265      // A + F + 1
#define K1_PAD  288      // layer-1 K padded to 9 chunks of 32
#define M_TILE  16

// ---- LDS layout (bytes), all 16B aligned ----
#define OFF_WT1   0                          // 256 x 288 f16  (W1^T, [N][K])
#define OFF_WT2   (OFF_WT1 + 256*K1_PAD*2)   // 256 x 256 f16  (W2^T)
#define OFF_WT3   (OFF_WT2 + 256*256*2)      // 16  x 256 f16  (W3^T, N padded 8->16)
#define OFF_X     (OFF_WT3 + 16*256*2)       // 16 x 288 f16   (input rows [a|c|t|pad])
#define OFF_XD    (OFF_X   + 16*K1_PAD*2)    // 16 x 32  f16   (tangent input, cols 8..31 = 0)
#define OFF_H     (OFF_XD  + 16*32*2)        // 16 x 256 f16   (hidden)
#define OFF_HD    (OFF_H   + 16*256*2)       // 16 x 256 f16   (hidden tangent)
#define OFF_AF    (OFF_HD  + 16*256*2)       // 16 x 8 f32     (current action, fp32 master)
#define OFF_DIV   (OFF_AF  + 16*8*4)         // 16 f32         (divergence integral)
#define OFF_B1    (OFF_DIV + 16*4)           // 256 f32
#define OFF_B2    (OFF_B1  + 256*4)          // 256 f32
#define OFF_B3    (OFF_B2  + 256*4)          // 8 f32
#define SMEM_BYTES (OFF_B3 + 32)

__device__ __forceinline__ v16h pack16(v8h lo, v8h hi) {
  v16h r;
#pragma unroll
  for (int i = 0; i < 8; ++i) { r[i] = lo[i]; r[i + 8] = hi[i]; }
  return r;
}

// A-operand (16x32 f16, row-major [M][ldk] in LDS).
// lane L: M = L%16; halves 0..7 -> K = k0 + off + 0..7, halves 8..15 -> K = k0+16+off+0..7
// with off = 8 for the upper half-wave (matches ISA 16-bit A layout).
__device__ __forceinline__ v16h ld_a_frag(const _Float16* rowmaj, int lane, int ldk, int k0) {
  int m   = lane & 15;
  int off = (lane & 16) ? 8 : 0;
  const _Float16* p = rowmaj + m * ldk + k0 + off;
  return pack16(*(const v8h*)p, *(const v8h*)(p + 16));
}

// B-operand (32x16 f16) from transposed weights [N][ldk] in LDS.
// lane L: N = nbase + L%16; halves 0..7 -> K = k0 + off + 0..7, halves 8..15 -> K = k0+16+off+0..7, off = 8 for upper half-wave.
__device__ __forceinline__ v16h ld_b_frag(const _Float16* colmaj, int lane, int ldk,
                                          int nbase, int k0) {
  int n   = nbase + (lane & 15);
  int off = (lane & 16) ? 8 : 0;
  const _Float16* p = colmaj + n * ldk + k0 + off;
  return pack16(*(const v8h*)p, *(const v8h*)(p + 16));
}

#define WMMA(a, b, c) \
  __builtin_amdgcn_wmma_f32_16x16x32_f16(false, (a), false, (b), (short)0, (c), false, false)

__global__ __launch_bounds__(256) void fm_logp_kernel(
    const float* __restrict__ actions, const float* __restrict__ feats,
    const float* __restrict__ W1, const float* __restrict__ b1g,
    const float* __restrict__ W2, const float* __restrict__ b2g,
    const float* __restrict__ W3, const float* __restrict__ b3g,
    const float* __restrict__ eps, float* __restrict__ out)
{
  extern __shared__ char smem[];
  _Float16* WT1 = (_Float16*)(smem + OFF_WT1);
  _Float16* WT2 = (_Float16*)(smem + OFF_WT2);
  _Float16* WT3 = (_Float16*)(smem + OFF_WT3);
  _Float16* x    = (_Float16*)(smem + OFF_X);
  _Float16* xd   = (_Float16*)(smem + OFF_XD);
  _Float16* hbuf = (_Float16*)(smem + OFF_H);
  _Float16* hdb  = (_Float16*)(smem + OFF_HD);
  float* a_f32 = (float*)(smem + OFF_AF);
  float* div_i = (float*)(smem + OFF_DIV);
  float* bb1   = (float*)(smem + OFF_B1);
  float* bb2   = (float*)(smem + OFF_B2);
  float* bb3   = (float*)(smem + OFF_B3);

  const int tid  = threadIdx.x;
  const int lane = tid & 31;
  const int wid  = tid >> 5;
  const int b0   = blockIdx.x * M_TILE;
  const float dt = 1.0f / (float)NSTEPS;

  // ---------- one-time setup ----------
  // W1^T (zero-pad K 265..287), k-major so global reads coalesce
  for (int idx = tid; idx < K1_PAD * 256; idx += 256) {
    int k = idx >> 8, n = idx & 255;
    float w = (k < IN_DIM) ? W1[(size_t)k * H_DIM + n] : 0.f;
    WT1[n * K1_PAD + k] = (_Float16)w;
  }
  for (int idx = tid; idx < 256 * 256; idx += 256) {
    int k = idx >> 8, n = idx & 255;
    WT2[n * 256 + k] = (_Float16)W2[(size_t)k * H_DIM + n];
  }
  for (int idx = tid; idx < 256 * 16; idx += 256) {
    int k = idx >> 4, n = idx & 15;
    float w = (n < A_DIM) ? W3[(size_t)k * A_DIM + n] : 0.f;
    WT3[n * 256 + k] = (_Float16)w;
  }
  // conditioning features into x cols 8..263
  for (int idx = tid; idx < M_TILE * 256; idx += 256) {
    int m = idx >> 8, j = idx & 255;
    x[m * K1_PAD + A_DIM + j] = (_Float16)feats[(size_t)(b0 + m) * F_DIM + j];
  }
  // zero x pad cols 265..287 and xd cols 8..31
  for (int idx = tid; idx < M_TILE * (K1_PAD - IN_DIM); idx += 256) {
    int m = idx / (K1_PAD - IN_DIM), q = idx % (K1_PAD - IN_DIM);
    x[m * K1_PAD + IN_DIM + q] = (_Float16)0.f;
  }
  for (int idx = tid; idx < M_TILE * 24; idx += 256) {
    int m = idx / 24, q = idx % 24;
    xd[m * 32 + 8 + q] = (_Float16)0.f;
  }
  if (tid < 128) {
    int m = tid >> 3, n = tid & 7;
    a_f32[m * 8 + n] = actions[(size_t)(b0 + m) * A_DIM + n];
  }
  if (tid < 16) div_i[tid] = 0.f;
  bb1[tid] = b1g[tid];
  bb2[tid] = b2g[tid];
  if (tid < 8) bb3[tid] = b3g[tid];
  __syncthreads();

  const int nb0 = (wid * 2) * 16;
  const int nb1 = nb0 + 16;

  // ---------- 50 Euler steps ----------
  for (int step = 0; step < NSTEPS; ++step) {
    const float tval = 1.0f - (float)step * dt;

    // phase 1: refresh a-columns, t-column, tangent probe
    if (tid < 128) {
      int m = tid >> 3, n = tid & 7;
      x[m * K1_PAD + n] = (_Float16)a_f32[m * 8 + n];
      xd[m * 32 + n] = (_Float16)eps[((size_t)step * B_TOT + b0 + m) * A_DIM + n];
    }
    if (tid >= 128 && tid < 128 + M_TILE)
      x[(tid - 128) * K1_PAD + (A_DIM + F_DIM)] = (_Float16)tval;
    __syncthreads();

    // ---- layer 1: z1 = x@W1, z1' = xd@W1 (tangent only touches K chunk 0) ----
    v8f f0 = {}, f1 = {}, t0 = {}, t1 = {};
#pragma unroll
    for (int kc = 0; kc < 9; ++kc) {
      int k0 = kc * 32;
      v16h af  = ld_a_frag(x, lane, K1_PAD, k0);
      v16h bf0 = ld_b_frag(WT1, lane, K1_PAD, nb0, k0);
      v16h bf1 = ld_b_frag(WT1, lane, K1_PAD, nb1, k0);
      f0 = WMMA(af, bf0, f0);
      f1 = WMMA(af, bf1, f1);
      if (kc == 0) {
        v16h at = ld_a_frag(xd, lane, 32, 0);
        t0 = WMMA(at, bf0, t0);
        t1 = WMMA(at, bf1, t1);
      }
    }
    {
      int nl = lane & 15, mh = (lane & 16) ? 8 : 0;
#pragma unroll
      for (int i = 0; i < 8; ++i) {
        int m = mh + i;
        float z  = f0[i] + bb1[nb0 + nl];
        float sg = 1.f / (1.f + __expf(-z));
        hbuf[m * 256 + nb0 + nl] = (_Float16)(z * sg);
        hdb [m * 256 + nb0 + nl] = (_Float16)(sg * (1.f + z * (1.f - sg)) * t0[i]);
        z  = f1[i] + bb1[nb1 + nl];
        sg = 1.f / (1.f + __expf(-z));
        hbuf[m * 256 + nb1 + nl] = (_Float16)(z * sg);
        hdb [m * 256 + nb1 + nl] = (_Float16)(sg * (1.f + z * (1.f - sg)) * t1[i]);
      }
    }
    __syncthreads();

    // ---- layer 2: z2 = h1@W2, z2' = h1'@W2 (shared B fragments) ----
    v8f g0 = {}, g1 = {}, u0 = {}, u1 = {};
#pragma unroll
    for (int kc = 0; kc < 8; ++kc) {
      int k0 = kc * 32;
      v16h ah  = ld_a_frag(hbuf, lane, 256, k0);
      v16h ad  = ld_a_frag(hdb,  lane, 256, k0);
      v16h bf0 = ld_b_frag(WT2, lane, 256, nb0, k0);
      v16h bf1 = ld_b_frag(WT2, lane, 256, nb1, k0);
      g0 = WMMA(ah, bf0, g0);
      u0 = WMMA(ad, bf0, u0);
      g1 = WMMA(ah, bf1, g1);
      u1 = WMMA(ad, bf1, u1);
    }
    __syncthreads();  // all waves finished reading h/h'
    {
      int nl = lane & 15, mh = (lane & 16) ? 8 : 0;
#pragma unroll
      for (int i = 0; i < 8; ++i) {
        int m = mh + i;
        float z  = g0[i] + bb2[nb0 + nl];
        float sg = 1.f / (1.f + __expf(-z));
        hbuf[m * 256 + nb0 + nl] = (_Float16)(z * sg);
        hdb [m * 256 + nb0 + nl] = (_Float16)(sg * (1.f + z * (1.f - sg)) * u0[i]);
        z  = g1[i] + bb2[nb1 + nl];
        sg = 1.f / (1.f + __expf(-z));
        hbuf[m * 256 + nb1 + nl] = (_Float16)(z * sg);
        hdb [m * 256 + nb1 + nl] = (_Float16)(sg * (1.f + z * (1.f - sg)) * u1[i]);
      }
    }
    __syncthreads();

    // ---- layer 3 head: wave0 -> v (Euler update), wave1 -> Jv (Hutchinson) ----
    if (wid == 0) {
      v8f acc = {};
#pragma unroll
      for (int kc = 0; kc < 8; ++kc) {
        int k0 = kc * 32;
        acc = WMMA(ld_a_frag(hbuf, lane, 256, k0),
                   ld_b_frag(WT3, lane, 256, 0, k0), acc);
      }
      int nl = lane & 15, mh = (lane & 16) ? 8 : 0;
      if (nl < A_DIM) {
#pragma unroll
        for (int i = 0; i < 8; ++i) {
          float v = acc[i] + bb3[nl];
          a_f32[(mh + i) * 8 + nl] -= v * dt;
        }
      }
    } else if (wid == 1) {
      v8f acc = {};
#pragma unroll
      for (int kc = 0; kc < 8; ++kc) {
        int k0 = kc * 32;
        acc = WMMA(ld_a_frag(hdb, lane, 256, k0),
                   ld_b_frag(WT3, lane, 256, 0, k0), acc);
      }
      int nl = lane & 15, mh = (lane & 16) ? 8 : 0;
#pragma unroll
      for (int i = 0; i < 8; ++i) {
        float ev = 0.f;
        if (nl < A_DIM)
          ev = eps[((size_t)step * B_TOT + b0 + mh + i) * A_DIM + nl];
        float p = acc[i] * ev;            // jv[M][n] * e[M][n]
        p += __shfl_xor(p, 1, 32);        // reduce over the 8 action columns
        p += __shfl_xor(p, 2, 32);
        p += __shfl_xor(p, 4, 32);
        if (nl == 0) div_i[mh + i] += p * dt;
      }
    }
    __syncthreads();
  }

  // ---------- epilogue: Gaussian log-prob at a0 minus divergence integral ----------
  if (tid < M_TILE) {
    float s = 0.f;
#pragma unroll
    for (int n = 0; n < 8; ++n) { float av = a_f32[tid * 8 + n]; s += av * av; }
    const float LOG2PI = 1.8378770664093453f;
    out[b0 + tid] = -0.5f * (s + (float)A_DIM * LOG2PI) - div_i[tid];
  }
}

extern "C" void kernel_launch(void* const* d_in, const int* in_sizes, int n_in,
                              void* d_out, int out_size, void* d_ws, size_t ws_size,
                              hipStream_t stream) {
  (void)in_sizes; (void)n_in; (void)out_size; (void)d_ws; (void)ws_size;
  const float* actions = (const float*)d_in[0];
  const float* feats   = (const float*)d_in[1];
  const float* W1      = (const float*)d_in[2];
  const float* b1      = (const float*)d_in[3];
  const float* W2      = (const float*)d_in[4];
  const float* b2      = (const float*)d_in[5];
  const float* W3      = (const float*)d_in[6];
  const float* b3      = (const float*)d_in[7];
  const float* eps     = (const float*)d_in[8];
  float* out = (float*)d_out;

  hipFuncSetAttribute((const void*)fm_logp_kernel,
                      hipFuncAttributeMaxDynamicSharedMemorySize, SMEM_BYTES);

  dim3 grid(B_TOT / M_TILE), block(256);
  fm_logp_kernel<<<grid, block, SMEM_BYTES, stream>>>(
      actions, feats, W1, b1, W2, b2, W3, b3, eps, out);
}
